// JointsGait_19602230739364
// MI455X (gfx1250) — matrix operand order repaired and checked by reading, not compiled, not run on hardware
//
#include <hip/hip_runtime.h>
#include <hip/hip_bf16.h>
#include <math.h>

// ---------------------------------------------------------------------------
// Types for CDNA5 WMMA (wave32): v_wmma_f32_16x16x32_bf16
// ---------------------------------------------------------------------------
typedef __bf16 bf16;
typedef __attribute__((ext_vector_type(16))) bf16  v16bf;
typedef __attribute__((ext_vector_type(8)))  bf16  v8bf;
typedef __attribute__((ext_vector_type(8)))  float v8f;

#define G_GRAPHS 10000
#define KPTS 17
#define GPB  4      // graphs per workgroup
#define ROWS 32     // padded rows per graph (17 real + 15 zero)
#define MAXD 256

// Layer dims. Weights are stored pre-staged+padded: [ks][Dout][32] bf16,
// K zero-padded to multiples of 32 (so Din=2 needs no special case).
constexpr int kDims[10] = {2, 64, 64, 64, 128, 128, 128, 256, 256, 256};
// padded sizes: Ksteps*Dout*32 halfs
constexpr int kWoff[9] = {0, 2048, 6144, 10240, 18432, 34816, 51200, 83968, 149504};

// JRPP pooling matrix P[6][17] (constant: means over fixed keypoint subsets)
#define S17 (1.0f/17.0f)
#define S11 (1.0f/11.0f)
#define S6  (1.0f/6.0f)
#define S5  (1.0f/5.0f)
constexpr float kP[6][KPTS] = {
  {S17,S17,S17,S17,S17,S17,S17,S17,S17,S17,S17,S17,S17,S17,S17,S17,S17},
  {S11,S11,S11,S11,S11,S11,S11,S11,S11,S11,S11, 0, 0, 0, 0, 0, 0},
  { 0, 0, 0, 0, 0, 0, 0, 0, 0, 0, 0, S6,S6,S6,S6,S6,S6},
  {S5,S5,S5,S5,S5, 0, 0, 0, 0, 0, 0, 0, 0, 0, 0, 0, 0},
  { 0, 0, 0, 0, 0, S6, 0, S6, 0, S6, 0, 0, S6, 0, S6, 0, S6},  // {5,7,9,12,14,16}
  { 0, 0, 0, 0, 0, 0, S6, 0, S6, 0, S6,S6, 0, S6, 0, S6, 0},   // {6,8,10,11,13,15}
};

// LDS carve (dynamic shared memory; ~147KB -> 2 workgroups/WGP on CDNA5)
#define OFF_H   0                     // bf16 [GPB][ROWS][MAXD]      = 65536 B
#define OFF_TB  65536                 // bf16 t^T [GPB][MAXD][ROWS]  = 65536 B
#define OFF_W   131072                // bf16 [256][32]              = 16384 B
#define OFF_AB  147456                // bf16 Ahat padded [32][32]   = 2048 B
#define OFF_ATT 149504                // f32  [GPB][ROWS]            = 512 B
#define OFF_RED 150016                // f32  [8]
#define SMEM_BYTES 150080

struct GcnParams {
  const float* x;
  const float* att;
  const float* bias[9];
  const float* fcW;
  const float* fcB;
  const float* bn_g;
  const float* bn_b;
  const float* bn_rm;
  const float* bn_rv;
  const float* Ahat;     // ws: 17x17 f32 normalized adjacency (with self loops)
  const bf16*  Wt;       // ws: bf16 weights, pre-staged [ks][Dout][32], K-padded
  float*       out;      // [G, 1536] f32
};

// ---------------------------------------------------------------------------
// Prep kernel 1: build Ahat[17][17] from graph 0's 38 edges (+self loops).
// Structure is identical for every graph, so one matrix serves all.
// ---------------------------------------------------------------------------
__global__ void prep_adj(const int* __restrict__ ei, int E, float* __restrict__ Ahat) {
  __shared__ float deg[KPTS];
  __shared__ float dinv[KPTS];
  const int t = threadIdx.x;
  if (t < KPTS) deg[t] = 1.0f;                 // self loop
  __syncthreads();
  if (t == 0) {
    for (int e = 0; e < 38; ++e) deg[ei[E + e]] += 1.0f;  // dst row
  }
  __syncthreads();
  if (t < KPTS) dinv[t] = rsqrtf(deg[t]);
  __syncthreads();
  for (int i = t; i < KPTS * KPTS; i += blockDim.x) Ahat[i] = 0.0f;
  __syncthreads();
  if (t == 0) {
    for (int i = 0; i < KPTS; ++i) Ahat[i * KPTS + i] = dinv[i] * dinv[i];
    for (int e = 0; e < 38; ++e) {
      const int s = ei[e], d = ei[E + e];
      Ahat[d * KPTS + s] += dinv[s] * dinv[d];
    }
  }
}

// ---------------------------------------------------------------------------
// Prep kernel 2: W [Din][Dout] f32 -> pre-staged bf16 blocks [ks][Dout][32],
// zero-padded along K so in-kernel staging is a raw contiguous copy.
// ---------------------------------------------------------------------------
__global__ void prep_wt(const float* __restrict__ W, bf16* __restrict__ Wt,
                        int Din, int Dout, int total) {
  const int i = blockIdx.x * blockDim.x + threadIdx.x;
  if (i < total) {
    const int k  = i & 31;
    const int n  = (i >> 5) % Dout;
    const int ks = i / (Dout * 32);
    const int kk = ks * 32 + k;
    Wt[i] = (kk < Din) ? (bf16)W[kk * Dout + n] : (bf16)0.0f;
  }
}

// ---------------------------------------------------------------------------
// One GCN layer, fully in LDS, both GEMMs on WMMA:
//   t  = h @ W            (K-tiled, weights staged in LDS)
//   h' = ReLU(att * (Ahat_pad @ t + bias))   (single K=32 WMMA, bias via C)
// NT = Dout/16 compile-time; wave w owns M-tile w (16 padded rows, graph w/2).
// ---------------------------------------------------------------------------
template <int NT>
__device__ __forceinline__ void run_layer(
    bf16* __restrict__ sH, bf16* __restrict__ sTb, bf16* __restrict__ sW,
    const bf16* __restrict__ sAb, const float* __restrict__ sAtt,
    const bf16* __restrict__ W, const float* __restrict__ bias, int Ksteps) {
  const int tid  = threadIdx.x;
  const int lane = tid & 31;
  const int wave = tid >> 5;
  const int Dout = NT * 16;

  v8f acc[NT];
#pragma unroll
  for (int n = 0; n < NT; ++n)
#pragma unroll
    for (int j = 0; j < 8; ++j) acc[n][j] = 0.0f;

  const int gg    = wave >> 1;          // graph within block
  const int rbase = (wave & 1) << 4;    // 0 or 16 within padded 32-row graph
  // A fragment addressing (contiguous k order, same as B): lanes 0-15 hold K{0..15},
  // lanes 16-31 hold K{16..31}, row = lane%16.
  const int arow = rbase + (lane & 15);
  const int ak0  = (lane >> 4) << 4;
  const int r0   = rbase + ((lane >> 4) << 3);   // C-layout first row for this lane
  const int cl   = lane & 15;                    // C-layout column-within-tile

  // ---- main GEMM: t = h @ W ----
  for (int ks = 0; ks < Ksteps; ++ks) {
    __syncthreads();                         // sW / sTb safe to overwrite
    {
      // Stage pre-padded W chunk: contiguous [Dout][32] bf16 = Dout*64 bytes.
      const float4* src4 = (const float4*)W + (size_t)ks * (Dout * 4);
      float4* dst4 = (float4*)sW;
      for (int c = tid; c < Dout * 4; c += 256) dst4[c] = src4[c];
      if (ks + 1 < Ksteps)                   // prefetch next K-chunk
        __builtin_prefetch((const void*)(src4 + Dout * 4 + tid), 0, 3);
    }
    __syncthreads();

    v16bf a;
    const bf16* hrow = sH + (gg * ROWS + arow) * MAXD + (ks << 5);
    *(float4*)&a         = *(const float4*)(hrow + ak0);
    *(((float4*)&a) + 1) = *(const float4*)(hrow + ak0 + 8);

#pragma unroll
    for (int n = 0; n < NT; ++n) {
      v16bf b;
      const bf16* wrow = sW + ((n * 16 + cl) << 5) + ((lane >> 4) << 4);
      *(float4*)&b         = *(const float4*)(wrow);
      *(((float4*)&b) + 1) = *(const float4*)(wrow + 8);
      acc[n] = __builtin_amdgcn_wmma_f32_16x16x32_bf16(
          false, a, false, b, (short)0, acc[n], false, false);
    }
  }

  // ---- spill t as bf16, TRANSPOSED: sTb[g][feat][row] ----
  // Each lane's 8 accumulator rows are contiguous -> one ds_store_b128.
#pragma unroll
  for (int n = 0; n < NT; ++n) {
    v8bf u;
#pragma unroll
    for (int v = 0; v < 8; ++v) u[v] = (bf16)acc[n][v];
    *(v8bf*)(sTb + ((gg * MAXD + n * 16 + cl) * ROWS + r0)) = u;
  }
  __syncthreads();

  // ---- aggregation GEMM: h' = ReLU(att * (Ahat_pad @ t + bias)) ----
  v16bf aA;
  const bf16* Arow = sAb + arow * 32;
  *(float4*)&aA         = *(const float4*)(Arow + ak0);
  *(((float4*)&aA) + 1) = *(const float4*)(Arow + ak0 + 8);

  float attv[8];
#pragma unroll
  for (int v = 0; v < 8; ++v) attv[v] = sAtt[gg * ROWS + r0 + v];  // 0 for pad rows

#pragma unroll
  for (int n = 0; n < NT; ++n) {
    const int c = n * 16 + cl;
    // B fragment: t^T gives 16 contiguous rows per lane.
    v16bf b;
    const bf16* tcol = sTb + (gg * MAXD + c) * ROWS + ((lane >> 4) << 4);
    *(float4*)&b         = *(const float4*)(tcol);
    *(((float4*)&b) + 1) = *(const float4*)(tcol + 8);
    // bias enters through the C operand (splat per column)
    const float bc = bias[c];
    v8f cb;
#pragma unroll
    for (int v = 0; v < 8; ++v) cb[v] = bc;
    const v8f agg = __builtin_amdgcn_wmma_f32_16x16x32_bf16(
        false, aA, false, b, (short)0, cb, false, false);
#pragma unroll
    for (int v = 0; v < 8; ++v) {
      const float hv = agg[v] * attv[v];     // att==0 keeps pad rows exactly 0
      sH[(gg * ROWS + r0 + v) * MAXD + c] = (bf16)(hv > 0.0f ? hv : 0.0f);
    }
  }
  __syncthreads();
}

// ---------------------------------------------------------------------------
// Fused kernel: 9 GCN layers + JRPP pooling + per-scale FC + BN + L2-normalize.
// One workgroup (8 waves) handles 4 graphs entirely in LDS (~147 KB).
// ---------------------------------------------------------------------------
__global__ void __launch_bounds__(256) gcn_fused(GcnParams p) {
  extern __shared__ char smem[];
  bf16*  sH   = (bf16*)(smem + OFF_H);
  bf16*  sTb  = (bf16*)(smem + OFF_TB);
  bf16*  sW   = (bf16*)(smem + OFF_W);
  bf16*  sAb  = (bf16*)(smem + OFF_AB);
  float* sAtt = (float*)(smem + OFF_ATT);
  float* sRed = (float*)(smem + OFF_RED);

  const int tid  = threadIdx.x;
  const int lane = tid & 31;
  const int wave = tid >> 5;
  const int g0   = blockIdx.x * GPB;

  // Zero padded h (vectorized)
  {
    float4 z4; z4.x = z4.y = z4.z = z4.w = 0.0f;
    float4* h4 = (float4*)sH;
    for (int i = tid; i < (GPB * ROWS * MAXD * 2) / 16; i += 256) h4[i] = z4;
  }
  // Ahat -> bf16, zero-padded to 32x32
  for (int i = tid; i < 32 * 32; i += 256) {
    const int r = i >> 5, c = i & 31;
    sAb[i] = (r < KPTS && c < KPTS) ? (bf16)p.Ahat[r * KPTS + c] : (bf16)0.0f;
  }
  // att staged per padded row (0 for pad rows)
  for (int i = tid; i < GPB * ROWS; i += 256) {
    const int g = i >> 5, r = i & 31;
    sAtt[i] = (r < KPTS) ? p.att[(size_t)(g0 + g) * KPTS + r] : 0.0f;
  }
  __syncthreads();
  // Load x (2 features) into rows 0..16
  for (int i = tid; i < GPB * KPTS * 2; i += 256) {
    const int f = i & 1;
    const int node = i >> 1;
    const int g = node / KPTS;
    const int r = node - g * KPTS;
    sH[(g * ROWS + r) * MAXD + f] =
        (bf16)p.x[((size_t)(g0 + g) * KPTS + r) * 2 + f];
  }
  __syncthreads();

#pragma unroll
  for (int l = 0; l < 9; ++l) {
    const int Din    = kDims[l];
    const int Dout   = kDims[l + 1];
    const int Ksteps = (Din + 31) >> 5;
    const bf16* W      = p.Wt + kWoff[l];
    const float* bias  = p.bias[l];
    if (Dout == 64)
      run_layer<4>(sH, sTb, sW, sAb, sAtt, W, bias, Ksteps);
    else if (Dout == 128)
      run_layer<8>(sH, sTb, sW, sAb, sAtt, W, bias, Ksteps);
    else
      run_layer<16>(sH, sTb, sW, sAb, sAtt, W, bias, Ksteps);
  }

  // ---- JRPP pooling: [17,256] -> [6,256] (constant matrix kP) ----
  float* sPool = (float*)sTb;        // [GPB][6][256] (reuse t buffer)
  float* sFt   = sPool + GPB * 6 * 256;
  for (int idx = tid; idx < GPB * 6 * 256; idx += 256) {
    const int f  = idx & 255;
    const int t2 = idx >> 8;
    const int s  = t2 % 6;
    const int g  = t2 / 6;
    const bf16* hb = sH + (g * ROWS) * MAXD + f;
    float v = 0.0f;
#pragma unroll
    for (int k = 0; k < KPTS; ++k) v += kP[s][k] * (float)hb[k * MAXD];
    sPool[idx] = v;
  }
  __syncthreads();

  // ---- per-scale FC + BN: one (s,o) pair per thread, ALL 4 graphs at once.
  // fcW row is loaded once and reused 4x; pooled reads are LDS broadcasts.
  for (int so = tid; so < 6 * 256; so += 256) {
    const int s = so >> 8;
    const int o = so & 255;
    const float4* w4 = (const float4*)(p.fcW + (size_t)so * 256);
    const float4* p0 = (const float4*)(sPool + (0 * 6 + s) * 256);
    const float4* p1 = (const float4*)(sPool + (1 * 6 + s) * 256);
    const float4* p2 = (const float4*)(sPool + (2 * 6 + s) * 256);
    const float4* p3 = (const float4*)(sPool + (3 * 6 + s) * 256);
    float v0 = 0.0f, v1 = 0.0f, v2 = 0.0f, v3 = 0.0f;
    for (int i = 0; i < 64; ++i) {
      const float4 w = w4[i];
      float4 a;
      a = p0[i]; v0 += w.x * a.x + w.y * a.y + w.z * a.z + w.w * a.w;
      a = p1[i]; v1 += w.x * a.x + w.y * a.y + w.z * a.z + w.w * a.w;
      a = p2[i]; v2 += w.x * a.x + w.y * a.y + w.z * a.z + w.w * a.w;
      a = p3[i]; v3 += w.x * a.x + w.y * a.y + w.z * a.z + w.w * a.w;
    }
    const float fb = p.fcB[so];
    const float sc = p.bn_g[so] * rsqrtf(p.bn_rv[so] + 1e-5f);
    const float sh = p.bn_b[so] - (p.bn_rm[so] - fb) * sc;  // (v+fb-rm)*sc+beta
    sFt[(0 * 6 + s) * 256 + o] = v0 * sc + sh;
    sFt[(1 * 6 + s) * 256 + o] = v1 * sc + sh;
    sFt[(2 * 6 + s) * 256 + o] = v2 * sc + sh;
    sFt[(3 * 6 + s) * 256 + o] = v3 * sc + sh;
  }
  __syncthreads();

  // ---- L2 normalize over 1536 per graph (wave g reduces graph g) ----
  if (wave < GPB) {
    float ss = 0.0f;
    const float* fb2 = sFt + wave * 1536;
    for (int j = lane; j < 1536; j += 32) { const float v = fb2[j]; ss += v * v; }
#pragma unroll
    for (int m = 16; m; m >>= 1) ss += __shfl_xor(ss, m, 32);
    if (lane == 0) sRed[wave] = 1.0f / fmaxf(sqrtf(ss), 1e-12f);
  }
  __syncthreads();

  for (int idx = tid; idx < GPB * 1536; idx += 256) {
    const int g = idx / 1536;
    p.out[(size_t)(g0 + g) * 1536 + (idx - g * 1536)] = sFt[idx] * sRed[g];
  }
}

// ---------------------------------------------------------------------------
// Launch
// ---------------------------------------------------------------------------
extern "C" void kernel_launch(void* const* d_in, const int* in_sizes, int n_in,
                              void* d_out, int out_size, void* d_ws, size_t ws_size,
                              hipStream_t stream) {
  (void)n_in; (void)out_size; (void)ws_size;

  const float* x   = (const float*)d_in[0];
  const float* att = (const float*)d_in[1];
  const int*   ei  = (const int*)d_in[2];
  const int    E   = in_sizes[2] / 2;

  float* Ahat = (float*)d_ws;
  bf16*  Wt   = (bf16*)((char*)d_ws + 4096);

  // Prep: adjacency + pre-staged bf16 weights (tiny; all L2-resident)
  prep_adj<<<1, 64, 0, stream>>>(ei, E, Ahat);
  for (int l = 0; l < 9; ++l) {
    const int Din = kDims[l], Dout = kDims[l + 1];
    const int Ksteps = (Din + 31) >> 5;
    const int total = Ksteps * Dout * 32;
    prep_wt<<<(total + 255) / 256, 256, 0, stream>>>(
        (const float*)d_in[3 + 2 * l], Wt + kWoff[l], Din, Dout, total);
  }

  GcnParams prm;
  prm.x = x;
  prm.att = att;
  for (int l = 0; l < 9; ++l) prm.bias[l] = (const float*)d_in[4 + 2 * l];
  prm.fcW   = (const float*)d_in[21];
  prm.fcB   = (const float*)d_in[22];
  prm.bn_g  = (const float*)d_in[23];
  prm.bn_b  = (const float*)d_in[24];
  prm.bn_rm = (const float*)d_in[25];
  prm.bn_rv = (const float*)d_in[26];
  prm.Ahat  = Ahat;
  prm.Wt    = Wt;
  prm.out   = (float*)d_out;

  (void)hipFuncSetAttribute((const void*)gcn_fused,
                            hipFuncAttributeMaxDynamicSharedMemorySize,
                            SMEM_BYTES);
  gcn_fused<<<G_GRAPHS / GPB, 256, SMEM_BYTES, stream>>>(prm);
}
